// SubnetworkEncoder_9216999817956
// MI455X (gfx1250) — matrix operand order repaired and checked by reading, not compiled, not run on hardware
//
#include <hip/hip_runtime.h>
#include <hip/hip_bf16.h>

// ---------------- types for WMMA ----------------
typedef __attribute__((ext_vector_type(16))) __bf16 v16bf;
typedef __attribute__((ext_vector_type(8)))  __bf16 v8bf;
typedef __attribute__((ext_vector_type(8)))  float  v8f;

#define ND_  40000
#define NDIS_ 40000
#define NP_  80000
#define NG_  80000
#define NPW_ 40000
#define D_   128

__device__ __forceinline__ __bf16 to_bf16(float f) {
    unsigned u = __builtin_bit_cast(unsigned, f);
    unsigned r = u + 0x7FFFu + ((u >> 16) & 1u);   // round-to-nearest-even
    unsigned short h = (unsigned short)(r >> 16);
    return __builtin_bit_cast(__bf16, h);
}

// ---------------- weight prep: Wt[m][n][k] = bf16(Wsrc[m][k][n]) ----------------
__global__ void prep_weights(const float* __restrict__ Wb, const float* __restrict__ W1,
                             __bf16* __restrict__ Wt) {
    int idx = blockIdx.x * 256 + threadIdx.x;          // 16 * 128 * 128
    if (idx >= 16 * 16384) return;
    int m = idx >> 14, rem = idx & 16383, n = rem >> 7, k = rem & 127;
    const float* src = (m < 15) ? (Wb + (size_t)m * 16384) : W1;
    Wt[idx] = to_bf16(src[k * 128 + n]);
}

// ---------------- degrees ----------------
__global__ void degrees_kernel(const int* __restrict__ src, const int* __restrict__ dst,
                               int E, float* __restrict__ degOut, float* __restrict__ degIn) {
    int i = blockIdx.x * 256 + threadIdx.x;
    if (i >= E) return;
    unsafeAtomicAdd(&degOut[src[i]], 1.0f);
    unsafeAtomicAdd(&degIn[dst[i]], 1.0f);
}

__global__ void finalize_deg(float* __restrict__ deg, int n) {
    int i = blockIdx.x * 256 + threadIdx.x;
    if (i >= n) return;
    float d = deg[i];
    deg[i] = rsqrtf(d < 1.0f ? 1.0f : d);
}

// ---------------- edge scatter: msg[dst] += x[src] * rdegOut[src] ----------------
__global__ void scatter_kernel(const float* __restrict__ X, const int* __restrict__ src,
                               const int* __restrict__ dst, const float* __restrict__ rdegOut,
                               float* __restrict__ msg, int E) {
    long long i = (long long)blockIdx.x * 256 + threadIdx.x;
    if (i >= (long long)E * 32) return;
    int e = (int)(i >> 5), l = (int)(i & 31);
    int s = src[e], d = dst[e];
    float sc = rdegOut[s];
    float4 v = *(const float4*)&X[(size_t)s * D_ + l * 4];
    float* p = &msg[(size_t)d * D_ + l * 4];
    unsafeAtomicAdd(p + 0, v.x * sc);
    unsafeAtomicAdd(p + 1, v.y * sc);
    unsafeAtomicAdd(p + 2, v.z * sc);
    unsafeAtomicAdd(p + 3, v.w * sc);
}

// ---------------- WMMA GEMM: Y (+)= (A * rowscale) @ Wt^T + bias ----------------
// block = 256 threads (8 waves); block covers 16 rows x 128 cols; wave w -> cols [16w,16w+16)
__global__ __launch_bounds__(256, 2)
void gemm_wmma(const float* __restrict__ A, const float* __restrict__ rowscale,
               const __bf16* __restrict__ Wt, const float* __restrict__ bias,
               float* __restrict__ Y, int M, int acc) {
    __shared__ __align__(16) __bf16 lA[16 * D_];
    int row0 = blockIdx.x * 16;
    int tid = threadIdx.x;
    // stage A tile (16x128 f32), fuse rsqrt(deg_in) row scale + bf16 convert
    for (int i = tid; i < 16 * D_; i += 256) {
        int r = i >> 7, c = i & 127;
        float s = rowscale ? rowscale[row0 + r] : 1.0f;
        lA[i] = to_bf16(A[(size_t)(row0 + r) * D_ + c] * s);
    }
    __syncthreads();

    int wave = tid >> 5, lane = tid & 31;
    int n0 = wave * 16;
    int hi = lane >> 4, lo = lane & 15;

    v8f c = {};
    if (acc) {
        for (int r = 0; r < 8; ++r)
            c[r] = Y[(size_t)(row0 + r + hi * 8) * D_ + n0 + lo];
    }
    for (int ks = 0; ks < 4; ++ks) {
        int kb = ks * 32;
        union { v16bf v; v8bf h[2]; } a, b;
        // A fragment (16-bit A 16x32 layout, ISA 7.12.2): lane -> row=lo,
        // VGPR0-3: K = hi*8 + 0..7 ; VGPR4-7: K = 16 + hi*8 + 0..7
        a.h[0] = *(const v8bf*)&lA[lo * D_ + kb + hi * 8];
        a.h[1] = *(const v8bf*)&lA[lo * D_ + kb + 16 + hi * 8];
        // B fragment: lane -> col=n0+lo, K = hi*8 + 0..7 and 16 + hi*8 + 0..7 (Wt is [N][K])
        const __bf16* bp = Wt + (size_t)(n0 + lo) * D_ + kb + hi * 8;
        b.h[0] = *(const v8bf*)bp;
        b.h[1] = *(const v8bf*)(bp + 16);
        c = __builtin_amdgcn_wmma_f32_16x16x32_bf16(false, a.v, false, b.v,
                                                    (short)0, c, false, false);
    }
    float bb = bias[n0 + lo];
    for (int r = 0; r < 8; ++r)
        Y[(size_t)(row0 + r + hi * 8) * D_ + n0 + lo] = c[r] + bb;
}

// ---------------- BatchNorm stats: per-column sum / sumsq ----------------
__global__ void bn_stats(const float* __restrict__ Y, float* __restrict__ sum,
                         float* __restrict__ sumsq, int M) {
    __shared__ float ls[256], ls2[256];
    int c = threadIdx.x & 127, h = threadIdx.x >> 7;
    int r0 = blockIdx.x * 256;
    float a = 0.f, a2 = 0.f;
    for (int r = r0 + h; r < r0 + 256 && r < M; r += 2) {
        float v = Y[(size_t)r * D_ + c];
        a += v; a2 += v * v;
    }
    ls[threadIdx.x] = a; ls2[threadIdx.x] = a2;
    __syncthreads();
    if (h == 0) {
        unsafeAtomicAdd(&sum[c],   ls[c]  + ls[c + 128]);
        unsafeAtomicAdd(&sumsq[c], ls2[c] + ls2[c + 128]);
    }
}

__global__ void bn_finalize(const float* __restrict__ sum, const float* __restrict__ sumsq,
                            const float* __restrict__ gamma, const float* __restrict__ beta,
                            float* __restrict__ scale, float* __restrict__ shift, int M) {
    int c = threadIdx.x;   // 128 threads
    float inv = 1.0f / (float)M;
    float mu = sum[c] * inv;
    float var = sumsq[c] * inv - mu * mu;
    float s = gamma[c] * rsqrtf(var + 1e-5f);
    scale[c] = s;
    shift[c] = beta[c] - mu * s;
}

__global__ void bn_apply(float* __restrict__ Y, const float* __restrict__ scale,
                         const float* __restrict__ shift, const float* __restrict__ a_ptr, int M) {
    int i = blockIdx.x * 256 + threadIdx.x;
    if (i >= M * D_) return;
    int c = i & 127;
    float a = a_ptr[0];
    float y = Y[i] * scale[c] + shift[c];
    Y[i] = y >= 0.f ? y : a * y;
}

// ---------------- attention score reduce: ssum += sum_n tanh(T[n,:]) . w2 ----------------
__global__ void score_reduce(const float* __restrict__ T, const float* __restrict__ w2,
                             float* __restrict__ ssum, int M) {
    int row = blockIdx.x * 8 + (threadIdx.x >> 5);
    int lane = threadIdx.x & 31;
    if (row >= M) return;
    float4 v = *(const float4*)&T[(size_t)row * D_ + lane * 4];
    float4 w = *(const float4*)&w2[lane * 4];
    float acc = tanhf(v.x) * w.x + tanhf(v.y) * w.y + tanhf(v.z) * w.z + tanhf(v.w) * w.w;
    for (int off = 16; off; off >>= 1) acc += __shfl_down(acc, off, 32);
    if (lane == 0) unsafeAtomicAdd(ssum, acc);
}

__global__ void compute_betas(const float* __restrict__ ssum, float* __restrict__ betas) {
    int g = threadIdx.x;
    if (g >= 5) return;
    const int   pa[5] = {0, 1, 3, 5, 7};
    const int   pb[5] = {2, 9, 4, 6, 8};
    const float Mg[5] = {40000.f, 40000.f, 80000.f, 80000.f, 40000.f};
    float m0 = ssum[pa[g]] / Mg[g];
    float m1 = ssum[pb[g]] / Mg[g];
    float mx = fmaxf(m0, m1);
    float e0 = expf(m0 - mx), e1 = expf(m1 - mx);
    float inv = 1.0f / (e0 + e1);
    betas[2 * g]     = e0 * inv;
    betas[2 * g + 1] = e1 * inv;
}

__global__ void combine(const float* __restrict__ z0, const float* __restrict__ z1,
                        const float* __restrict__ bet, float* __restrict__ o, int n) {
    int i = blockIdx.x * 256 + threadIdx.x;
    if (i >= n) return;
    o[i] = bet[0] * z0[i] + bet[1] * z1[i];
}

// ---------------- host orchestration ----------------
extern "C" void kernel_launch(void* const* d_in, const int* in_sizes, int n_in,
                              void* d_out, int out_size, void* d_ws, size_t ws_size,
                              hipStream_t stream) {
    const float* h[5];
    for (int t = 0; t < 5; ++t) h[t] = (const float*)d_in[t];
    const int* esrc[10]; const int* edst[10]; int E[10];
    for (int r = 0; r < 10; ++r) {
        esrc[r] = (const int*)d_in[5 + 2 * r];
        edst[r] = (const int*)d_in[6 + 2 * r];
        E[r]    = in_sizes[5 + 2 * r];
    }
    const float* W_blocks = (const float*)d_in[25];
    const float* b_blocks = (const float*)d_in[26];
    const float* bn_gamma = (const float*)d_in[27];
    const float* bn_beta  = (const float*)d_in[28];
    const float* prelu_a  = (const float*)d_in[29];
    const float* attn_W1  = (const float*)d_in[30];
    const float* attn_b1  = (const float*)d_in[31];
    const float* attn_w2  = (const float*)d_in[32];
    float* out = (float*)d_out;

    // relation meta: srcType, n_src, n_dst  (dd,ddi,disdis,dp,pp,pg,gg,gpw,pwpw,pwdis)
    static const int srcType[10] = {0, 0, 1, 0, 2, 2, 3, 3, 4, 4};
    static const int nsrc[10]    = {ND_, ND_, NDIS_, ND_, NP_, NP_, NG_, NG_, NPW_, NPW_};
    static const int ndst[10]    = {ND_, NDIS_, NDIS_, NP_, NP_, NG_, NG_, NPW_, NPW_, NDIS_};
    // act-output buffers: d0,dis0,d1,p1,p2,g2,g3,pw3,pw4,dis4
    static const int rowsY[10]   = {ND_, NDIS_, ND_, NP_, NP_, NG_, NG_, NPW_, NPW_, NDIS_};
    static const int blkOf[10]   = {0, 0, 1, 1, 2, 2, 3, 3, 4, 4};

    // ---- workspace layout (float units, 256B aligned chunks) ----
    size_t off = 0;
    auto alloc = [&](size_t nf) { size_t o = off; off += (nf + 63) & ~(size_t)63; return o; };
    size_t degOutOff[10], degInOff[10], msgOff[10], YOff[10];
    size_t degBase = off;
    for (int r = 0; r < 10; ++r) { degOutOff[r] = alloc(nsrc[r]); degInOff[r] = alloc(ndst[r]); }
    size_t degEnd = off;
    for (int r = 0; r < 10; ++r) msgOff[r] = alloc((size_t)ndst[r] * D_);
    size_t msgBase = msgOff[0], msgEnd = off;
    for (int b = 0; b < 10; ++b) YOff[b] = alloc((size_t)rowsY[b] * D_);
    size_t statsOff = alloc(10 * 512);   // per buffer: sum[128] sumsq[128] scale[128] shift[128]
    size_t ssumOff  = alloc(64);
    size_t betasOff = alloc(64);
    size_t wtOff    = alloc(16 * 8192);  // 16 matrices of 128x128 bf16 (2 bf16 per float)
    float*  W  = (float*)d_ws;
    __bf16* Wt = (__bf16*)(W + wtOff);

    // ---- zero accumulators ----
    hipMemsetAsync(W + degBase, 0, (degEnd - degBase) * sizeof(float), stream);
    hipMemsetAsync(W + msgBase, 0, (msgEnd - msgBase) * sizeof(float), stream);
    hipMemsetAsync(W + statsOff, 0, 10 * 512 * sizeof(float), stream);
    hipMemsetAsync(W + ssumOff, 0, 64 * sizeof(float), stream);

    // ---- weights -> bf16 transposed ----
    prep_weights<<<(16 * 16384 + 255) / 256, 256, 0, stream>>>(W_blocks, attn_W1, Wt);

    // ---- degrees ----
    for (int r = 0; r < 10; ++r)
        degrees_kernel<<<(E[r] + 255) / 256, 256, 0, stream>>>(
            esrc[r], edst[r], E[r], W + degOutOff[r], W + degInOff[r]);
    finalize_deg<<<((int)(degEnd - degBase) + 255) / 256, 256, 0, stream>>>(
        W + degBase, (int)(degEnd - degBase));

    // ---- shared per-relation aggregation (10 scatters cover all 15 GCN calls) ----
    for (int r = 0; r < 10; ++r) {
        long long th = (long long)E[r] * 32;
        scatter_kernel<<<(unsigned)((th + 255) / 256), 256, 0, stream>>>(
            h[srcType[r]], esrc[r], edst[r], W + degOutOff[r], W + msgOff[r], E[r]);
    }

    // ---- 15 GCN GEMMs: {Ybuf, rel, weightIdx, acc} ----
    static const int gs[15][4] = {
        {0, 0,  0, 0}, {1, 1,  1, 0}, {1, 2,  2, 1}, {2, 0,  3, 0}, {3, 3,  4, 0},
        {3, 4,  5, 1}, {4, 4,  6, 0}, {5, 5,  7, 0}, {5, 6,  8, 1}, {6, 6,  9, 0},
        {7, 7, 10, 0}, {7, 8, 11, 1}, {8, 8, 12, 0}, {9, 9, 13, 0}, {9, 2, 14, 1},
    };
    for (int i = 0; i < 15; ++i) {
        int b = gs[i][0], r = gs[i][1], w = gs[i][2], acc = gs[i][3];
        gemm_wmma<<<rowsY[b] / 16, 256, 0, stream>>>(
            W + msgOff[r], W + degInOff[r], Wt + (size_t)w * 16384,
            b_blocks + (size_t)w * 128, W + YOff[b], rowsY[b], acc);
    }

    // ---- BatchNorm + PReLU (in place) ----
    for (int b = 0; b < 10; ++b) {
        int M = rowsY[b], blk = blkOf[b];
        float* st = W + statsOff + (size_t)b * 512;
        bn_stats<<<(M + 255) / 256, 256, 0, stream>>>(W + YOff[b], st, st + 128, M);
        bn_finalize<<<1, 128, 0, stream>>>(st, st + 128, bn_gamma + blk * 128,
                                           bn_beta + blk * 128, st + 256, st + 384, M);
        bn_apply<<<(M * D_ + 255) / 256, 256, 0, stream>>>(
            W + YOff[b], st + 256, st + 384, prelu_a + blk, M);
    }

    // ---- semantic attention: scores via WMMA GEMM into scratch, reduce to per-branch mean ----
    for (int b = 0; b < 10; ++b) {
        int M = rowsY[b];
        gemm_wmma<<<M / 16, 256, 0, stream>>>(
            W + YOff[b], nullptr, Wt + (size_t)15 * 16384, attn_b1, W + msgBase, M, 0);
        score_reduce<<<M / 8, 256, 0, stream>>>(W + msgBase, attn_w2, W + ssumOff + b, M);
    }
    compute_betas<<<1, 32, 0, stream>>>(W + ssumOff, W + betasOff);

    // ---- weighted combine into d_out (drug, disease, protein, gene, pathway) ----
    static const int    pa[5]  = {0, 1, 3, 5, 7};
    static const int    pb[5]  = {2, 9, 4, 6, 8};
    static const int    gn[5]  = {ND_, NDIS_, NP_, NG_, NPW_};
    static const size_t oOf[5] = {0, (size_t)ND_ * D_, (size_t)(ND_ + NDIS_) * D_,
                                  (size_t)(ND_ + NDIS_ + NP_) * D_,
                                  (size_t)(ND_ + NDIS_ + NP_ + NG_) * D_};
    for (int g = 0; g < 5; ++g) {
        int n = gn[g] * D_;
        combine<<<(n + 255) / 256, 256, 0, stream>>>(
            W + YOff[pa[g]], W + YOff[pb[g]], W + betasOff + 2 * g, out + oOf[g], n);
    }
}
